// Attention_11579231830842
// MI455X (gfx1250) — matrix operand order repaired and checked by reading, not compiled, not run on hardware
//
#include <hip/hip_runtime.h>

typedef __attribute__((ext_vector_type(16))) _Float16 v16h;
typedef __attribute__((ext_vector_type(8)))  _Float16 v8h;
typedef __attribute__((ext_vector_type(8)))  float    v8f;

#define B_      4
#define NQ_     1024
#define NK_     1024
#define H_      8
#define HD_     64
#define DM_     512      // d_model
#define SCALE_  0.125f   // 64^-0.5

__device__ __forceinline__ v8f wmma_f16(v16h a, v16h b, v8f c) {
  // (neg_a, A, neg_b, B, c_mod, C, reuse_a, reuse_b)
  return __builtin_amdgcn_wmma_f32_16x16x32_f16(false, a, false, b, (short)0, c, false, false);
}

// ---- CDNA5 async copy: 16B/lane global -> LDS, tracked by ASYNCcnt -----
__device__ __forceinline__ void async_b128(uint32_t lds_off, const void* gptr) {
  asm volatile("global_load_async_to_lds_b128 %0, %1, off"
               :: "v"(lds_off), "v"((unsigned long long)(size_t)gptr)
               : "memory");
}
__device__ __forceinline__ void wait_async0() {
#if __has_builtin(__builtin_amdgcn_s_wait_asynccnt)
  __builtin_amdgcn_s_wait_asynccnt(0);
#else
  asm volatile("s_wait_asynccnt 0x0" ::: "memory");
#endif
}

// ---- fragment loaders --------------------------------------------------
// A (16x32 f16): lane m=lane&15, hf=lane>>4 holds K = hf*8 + [0..8) and 16 + hf*8 + [0..8)
__device__ __forceinline__ v16h load_a_f32(const float* src, int ld, int row0, int k0, int lane) {
  int m = lane & 15, hf = lane >> 4;
  const float* p = src + (size_t)(row0 + m) * ld + k0 + hf * 8;
  v8f lo = *(const v8f*)p;
  v8f hi = *(const v8f*)(p + 16);
  v16h a;
#pragma unroll
  for (int i = 0; i < 8; ++i) { a[i] = (_Float16)lo[i]; a[i + 8] = (_Float16)hi[i]; }
  return a;
}

__device__ __forceinline__ v16h load_a_f16(const _Float16* src, int ld, int row0, int k0, int lane) {
  int m = lane & 15, hf = lane >> 4;
  const _Float16* p = src + (size_t)(row0 + m) * ld + k0 + hf * 8;
  v8h lo = *(const v8h*)p;
  v8h hi = *(const v8h*)(p + 16);
  v16h a;
#pragma unroll
  for (int i = 0; i < 8; ++i) { a[i] = lo[i]; a[i + 8] = hi[i]; }
  return a;
}

// B (32x16 f16) from an N-major [N][K] matrix: lane n=lane&15, hf=lane>>4 holds
// K = k0 + hf*8 + [0..8) and k0 + 16 + hf*8 + [0..8) for column col0+n
__device__ __forceinline__ v16h load_b_nmaj(const _Float16* src, int ld, int k0, int col0, int lane) {
  int n = lane & 15, hf = lane >> 4;
  const _Float16* p = src + (size_t)(col0 + n) * ld + k0 + hf * 8;
  v8h lo = *(const v8h*)p;
  v8h hi = *(const v8h*)(p + 16);
  v16h b;
#pragma unroll
  for (int i = 0; i < 8; ++i) { b[i] = lo[i]; b[i + 8] = hi[i]; }
  return b;
}

// ---- kernel 0: weight f32[K][N] -> f16[N][K] ---------------------------
__global__ __launch_bounds__(256) void wconv_kernel(const float* __restrict__ w,
                                                    _Float16* __restrict__ wt,
                                                    int K, int N) {
  int idx = blockIdx.x * 256 + threadIdx.x;
  if (idx < K * N) {
    int k = idx / N, n = idx % N;
    wt[(size_t)n * K + k] = (_Float16)w[idx];
  }
}

// ---- kernel 1: QKV projection GEMM (f32 A, f16 Bt), per-head scatter ---
// All 8 waves of a block share one B tile (64 cols x 32 k): async-staged in
// LDS, double buffered.  mode 0: Q scatter; mode 1: K + transposed-V scatter.
__global__ __launch_bounds__(256) void proj_qkv_kernel(const float* __restrict__ A,
    const _Float16* __restrict__ Bt, _Float16* __restrict__ Qh,
    _Float16* __restrict__ Kh, _Float16* __restrict__ Vt, int mode) {
  __shared__ __align__(64) _Float16 Blds[2][64 * 32];   // 4KB per buffer

  int tid = threadIdx.x;
  int lane = tid & 31, wv = tid >> 5;
  int wid = blockIdx.x * 8 + wv;
  const int tilesM = (B_ * NQ_) / 32;        // 128 -> 8 waves: consecutive tm, same tn
  int tm = wid % tilesM, tn = wid / tilesM;
  int m0 = tm * 32, n0 = tn * 64;

  int br = tid >> 2, bc = tid & 3;           // 64 rows x 4 x 16B chunks = 4KB tile

  // prologue: stage first B tile
  async_b128((uint32_t)(size_t)&Blds[0][br * 32 + bc * 8],
             Bt + (size_t)(n0 + br) * DM_ + bc * 8);
  wait_async0();
  __syncthreads();

  v8f acc[2][4] = {};
  int cur = 0;
  for (int kk = 0; kk < DM_; kk += 32) {
    if (kk + 32 < DM_)                        // prefetch next B tile into other buffer
      async_b128((uint32_t)(size_t)&Blds[cur ^ 1][br * 32 + bc * 8],
                 Bt + (size_t)(n0 + br) * DM_ + (kk + 32) + bc * 8);
    v16h a0 = load_a_f32(A, DM_, m0, kk, lane);
    v16h a1 = load_a_f32(A, DM_, m0 + 16, kk, lane);
#pragma unroll
    for (int j = 0; j < 4; ++j) {
      v16h bf = load_b_nmaj(&Blds[cur][0], 32, 0, j * 16, lane);
      acc[0][j] = wmma_f16(a0, bf, acc[0][j]);
      acc[1][j] = wmma_f16(a1, bf, acc[1][j]);
    }
    wait_async0();
    __syncthreads();
    cur ^= 1;
  }

  int n16 = lane & 15, hf = lane >> 4;
#pragma unroll
  for (int i = 0; i < 2; ++i)
#pragma unroll
    for (int j = 0; j < 4; ++j)
#pragma unroll
      for (int g = 0; g < 8; ++g) {
        int m = m0 + i * 16 + g + 8 * hf;    // C layout: M = g + 8*(lane/16)
        int n = n0 + j * 16 + n16;           // N = lane%16
        float v = acc[i][j][g];
        int bidx = m >> 10, row = m & 1023;
        if (mode == 0) {
          int h = n >> 6, d = n & 63;
          Qh[(((size_t)bidx * H_ + h) * NQ_ + row) * HD_ + d] = (_Float16)v;
        } else if (n < DM_) {
          int h = n >> 6, d = n & 63;
          Kh[(((size_t)bidx * H_ + h) * NK_ + row) * HD_ + d] = (_Float16)v;
        } else {
          int n2 = n - DM_;
          int h = n2 >> 6, d = n2 & 63;
          Vt[(((size_t)bidx * H_ + h) * HD_ + d) * NK_ + row] = (_Float16)v;
        }
      }
}

// ---- kernel 2: attention. 1 block = (b,h, 16 q rows); wave wv owns 128 keys
__global__ __launch_bounds__(256) void attn_kernel(
    const _Float16* __restrict__ Qh, const _Float16* __restrict__ Kh,
    const _Float16* __restrict__ Vt, float* __restrict__ attn_out,
    _Float16* __restrict__ Of) {
  __shared__ __align__(64) _Float16 Qlds[16 * 64];      // shared Q tile (2KB)
  __shared__ __align__(64) _Float16 Plds[8][16 * 128];  // per-wave P relay (32KB)
  __shared__ float redmax[8][16];
  __shared__ float redsum[8][16];
  __shared__ float Osh[16][64];

  int tid = threadIdx.x;
  int lane = tid & 31, wv = tid >> 5;
  int n16 = lane & 15, hf = lane >> 4;

  int qt = blockIdx.x & 63;                // NQ/16 = 64
  int bh = blockIdx.x >> 6;
  int h = bh & 7, b = bh >> 3;
  int q0 = qt * 16;

  const _Float16* Qbh = Qh + ((size_t)(b * H_ + h)) * NQ_ * HD_;
  const _Float16* Kbh = Kh + ((size_t)(b * H_ + h)) * NK_ * HD_;
  const _Float16* Vbh = Vt + ((size_t)(b * H_ + h)) * HD_ * NK_;
  float* attn_bh = attn_out + (((size_t)(b * H_ + h)) * NQ_ + q0) * NK_;

  for (int i = tid; i < 16 * 64; i += 256) Osh[i >> 6][i & 63] = 0.0f;

  // async-stage the Q tile (all 8 waves consume the same 16x64 block)
  if (tid < 128) {
    int r = tid >> 3, c = tid & 7;         // 16 rows x 8 x 16B chunks
    async_b128((uint32_t)(size_t)&Qlds[r * 64 + c * 8],
               Qbh + (size_t)(q0 + r) * HD_ + c * 8);
  }
  wait_async0();
  __syncthreads();

  int k0 = wv * 128;

  // ---- S = (Q K^T) * scale : 16 WMMAs per wave
  v16h aq0 = load_a_f16(Qlds, HD_, 0, 0, lane);
  v16h aq1 = load_a_f16(Qlds, HD_, 0, 32, lane);
  v8f acc[8];
#pragma unroll
  for (int t = 0; t < 8; ++t) {
    v8f c = {};
    c = wmma_f16(aq0, load_b_nmaj(Kbh, HD_, 0, k0 + t * 16, lane), c);
    c = wmma_f16(aq1, load_b_nmaj(Kbh, HD_, 32, k0 + t * 16, lane), c);
    acc[t] = c;
  }
#pragma unroll
  for (int t = 0; t < 8; ++t)
#pragma unroll
    for (int g = 0; g < 8; ++g) acc[t][g] *= SCALE_;

  // ---- row max: lane-local over 8 tiles, shfl over 16-lane half, LDS over waves
  float mrow[8];
#pragma unroll
  for (int g = 0; g < 8; ++g) {
    float mv = acc[0][g];
#pragma unroll
    for (int t = 1; t < 8; ++t) mv = fmaxf(mv, acc[t][g]);
    mrow[g] = mv;
  }
#pragma unroll
  for (int off = 1; off < 16; off <<= 1)
#pragma unroll
    for (int g = 0; g < 8; ++g) mrow[g] = fmaxf(mrow[g], __shfl_xor(mrow[g], off, 32));
  if (n16 == 0)
#pragma unroll
    for (int g = 0; g < 8; ++g) redmax[wv][hf * 8 + g] = mrow[g];
  __syncthreads();
#pragma unroll
  for (int g = 0; g < 8; ++g) {
    float mv = redmax[0][hf * 8 + g];
#pragma unroll
    for (int w2 = 1; w2 < 8; ++w2) mv = fmaxf(mv, redmax[w2][hf * 8 + g]);
    mrow[g] = mv;
  }

  // ---- exp + row sum
  float srow[8] = {0.f, 0.f, 0.f, 0.f, 0.f, 0.f, 0.f, 0.f};
#pragma unroll
  for (int t = 0; t < 8; ++t)
#pragma unroll
    for (int g = 0; g < 8; ++g) {
      float e = __expf(acc[t][g] - mrow[g]);
      acc[t][g] = e;
      srow[g] += e;
    }
#pragma unroll
  for (int off = 1; off < 16; off <<= 1)
#pragma unroll
    for (int g = 0; g < 8; ++g) srow[g] += __shfl_xor(srow[g], off, 32);
  if (n16 == 0)
#pragma unroll
    for (int g = 0; g < 8; ++g) redsum[wv][hf * 8 + g] = srow[g];
  __syncthreads();
#pragma unroll
  for (int g = 0; g < 8; ++g) {
    float sv = 0.0f;
#pragma unroll
    for (int w2 = 0; w2 < 8; ++w2) sv += redsum[w2][hf * 8 + g];
    srow[g] = __builtin_amdgcn_rcpf(sv);
  }

  // ---- P: NT-store to attn output (134MB, never re-read), stage f16 in LDS
#pragma unroll
  for (int t = 0; t < 8; ++t)
#pragma unroll
    for (int g = 0; g < 8; ++g) {
      int r = hf * 8 + g;
      float p = acc[t][g] * srow[g];
      __builtin_nontemporal_store(p, &attn_bh[(size_t)r * NK_ + k0 + t * 16 + n16]);
      Plds[wv][r * 128 + t * 16 + n16] = (_Float16)p;
    }

  // ---- O_partial = P V : 16 WMMAs per wave (A from own LDS region, no barrier needed)
  v8f oacc[4] = {};
#pragma unroll
  for (int ks = 0; ks < 4; ++ks) {
    v16h ap = load_a_f16(&Plds[wv][0], 128, 0, ks * 32, lane);
#pragma unroll
    for (int j = 0; j < 4; ++j)
      oacc[j] = wmma_f16(ap, load_b_nmaj(Vbh, NK_, k0 + ks * 32, j * 16, lane), oacc[j]);
  }
  // combine the 8 waves' partials
#pragma unroll
  for (int j = 0; j < 4; ++j)
#pragma unroll
    for (int g = 0; g < 8; ++g)
      atomicAdd(&Osh[hf * 8 + g][j * 16 + n16], oacc[j][g]);
  __syncthreads();

  // ---- Of[b][q][h*64+d] (bqhd layout for final projection)
  for (int i = tid; i < 16 * 64; i += 256) {
    int r = i >> 6, d = i & 63;
    Of[((size_t)b * NQ_ + q0 + r) * DM_ + h * HD_ + d] = (_Float16)Osh[r][d];
  }
}

// ---- kernel 3: out = Of @ w_proj + b_proj ------------------------------
__global__ __launch_bounds__(256) void proj_out_kernel(const _Float16* __restrict__ Of,
    const _Float16* __restrict__ Wpt, const float* __restrict__ bias,
    float* __restrict__ out) {
  __shared__ __align__(64) _Float16 Blds[2][64 * 32];

  int tid = threadIdx.x;
  int lane = tid & 31, wv = tid >> 5;
  int wid = blockIdx.x * 8 + wv;
  const int tilesM = (B_ * NQ_) / 32;
  int tm = wid % tilesM, tn = wid / tilesM;
  int m0 = tm * 32, n0 = tn * 64;

  int br = tid >> 2, bc = tid & 3;

  async_b128((uint32_t)(size_t)&Blds[0][br * 32 + bc * 8],
             Wpt + (size_t)(n0 + br) * DM_ + bc * 8);
  wait_async0();
  __syncthreads();

  v8f acc[2][4] = {};
  int cur = 0;
  for (int kk = 0; kk < DM_; kk += 32) {
    if (kk + 32 < DM_)
      async_b128((uint32_t)(size_t)&Blds[cur ^ 1][br * 32 + bc * 8],
                 Wpt + (size_t)(n0 + br) * DM_ + (kk + 32) + bc * 8);
    v16h a0 = load_a_f16(Of, DM_, m0, kk, lane);
    v16h a1 = load_a_f16(Of, DM_, m0 + 16, kk, lane);
#pragma unroll
    for (int j = 0; j < 4; ++j) {
      v16h bf = load_b_nmaj(&Blds[cur][0], 32, 0, j * 16, lane);
      acc[0][j] = wmma_f16(a0, bf, acc[0][j]);
      acc[1][j] = wmma_f16(a1, bf, acc[1][j]);
    }
    wait_async0();
    __syncthreads();
    cur ^= 1;
  }

  int n16 = lane & 15, hf = lane >> 4;
#pragma unroll
  for (int i = 0; i < 2; ++i)
#pragma unroll
    for (int j = 0; j < 4; ++j)
#pragma unroll
      for (int g = 0; g < 8; ++g) {
        int m = m0 + i * 16 + g + 8 * hf;
        int n = n0 + j * 16 + n16;
        out[(size_t)m * DM_ + n] = acc[i][j][g] + bias[n];
      }
}

// ---- host launch -------------------------------------------------------
extern "C" void kernel_launch(void* const* d_in, const int* in_sizes, int n_in,
                              void* d_out, int out_size, void* d_ws, size_t ws_size,
                              hipStream_t stream) {
  const float* x      = (const float*)d_in[0];   // [4,1024,512]
  const float* x_q    = (const float*)d_in[1];   // [4,1024,512]
  const float* w_q    = (const float*)d_in[2];   // [512,512]
  const float* w_kv   = (const float*)d_in[3];   // [512,1024]
  const float* w_proj = (const float*)d_in[4];   // [512,512]
  const float* b_proj = (const float*)d_in[5];   // [512]

  float* out_ptr  = (float*)d_out;                           // [4,1024,512]
  float* attn_ptr = (float*)d_out + (size_t)B_ * NQ_ * DM_;  // [4,8,1024,1024]

  const size_t T = (size_t)B_ * NQ_ * DM_;  // 2,097,152 halves per tensor
  _Float16* Qh   = (_Float16*)d_ws;
  _Float16* Kh   = Qh + T;
  _Float16* Vt   = Kh + T;
  _Float16* Of   = Vt + T;
  _Float16* Wqt  = Of + T;                  // 512*512
  _Float16* Wkvt = Wqt + 512 * 512;         // 1024*512
  _Float16* Wpt  = Wkvt + 1024 * 512;       // 512*512

  wconv_kernel<<<(512 * 512 + 255) / 256, 256, 0, stream>>>(w_q, Wqt, 512, 512);
  wconv_kernel<<<(512 * 1024 + 255) / 256, 256, 0, stream>>>(w_kv, Wkvt, 512, 1024);
  wconv_kernel<<<(512 * 512 + 255) / 256, 256, 0, stream>>>(w_proj, Wpt, 512, 512);

  // GEMM1: 4096x512x512 -> Q.  waves = 128*8 = 1024 -> 128 blocks
  proj_qkv_kernel<<<128, 256, 0, stream>>>(x_q, Wqt, Qh, Kh, Vt, 0);
  // GEMM2: 4096x1024x512 -> K,V. waves = 128*16 = 2048 -> 256 blocks
  proj_qkv_kernel<<<256, 256, 0, stream>>>(x, Wkvt, Qh, Kh, Vt, 1);

  // attention: B*H*(NQ/16) = 2048 blocks
  attn_kernel<<<2048, 256, 0, stream>>>(Qh, Kh, Vt, attn_ptr, Of);

  // output projection + bias
  proj_out_kernel<<<128, 256, 0, stream>>>(Of, Wpt, b_proj, out_ptr);
}
